// NequIPLayer_13950053777578
// MI455X (gfx1250) — matrix operand order repaired and checked by reading, not compiled, not run on hardware
//
#include <hip/hip_runtime.h>

typedef __attribute__((ext_vector_type(16))) _Float16 v16h;
typedef __attribute__((ext_vector_type(8)))  _Float16 v8h;
typedef __attribute__((ext_vector_type(8)))  float    v8f;

#define HID   128
#define NN    512
#define NNODE 1024   // B*N = 2*512

__device__ __forceinline__ float silu_f(float x) { return x / (1.0f + __expf(-x)); }

__device__ __forceinline__ v16h cat16(v8h lo, v8h hi) {
  return __builtin_shufflevector(lo, hi, 0,1,2,3,4,5,6,7,8,9,10,11,12,13,14,15);
}

// A-fragment (16x32 f16, M=rows) gathered from a row-major buffer.
// lanes 0-15 hold K kb+0..7 then kb+16..23 with kb=0; lanes 16-31 same with kb=8.
__device__ __forceinline__ v16h load_a16(const _Float16* base, int strideH, int row,
                                         int k0, int lane) {
  const int kb = (lane >> 4) * 8;
  const _Float16* p = base + row * strideH + k0 + kb;
  v8h lo = *(const v8h*)p;
  v8h hi = *(const v8h*)(p + 16);
  return cat16(lo, hi);
}

// B-fragment from pre-swizzled weights: 512 halfs per fragment, 16 contiguous per lane.
__device__ __forceinline__ v16h load_bfrag(const _Float16* sw, int fragIdx, int lane) {
  const v8h* p = (const v8h*)(sw + fragIdx * 512 + lane * 16);
  return cat16(p[0], p[1]);
}

// ---------------- prep kernels ----------------

// Pre-swizzle a [K x N] row-major f32 weight into WMMA B-fragment order (f16).
// B layout (32x16): lane column n = lane&15; halfs j=0..7 -> k = 8*(lane>>4) + j, j=8..15 -> k = 16 + 8*(lane>>4) + (j-8).
__global__ void swizzle_w16(const float* __restrict__ src, _Float16* __restrict__ dst,
                            int K, int N, int Kvalid) {
  int tid = blockIdx.x * blockDim.x + threadIdx.x;
  int total = (K >> 5) * (N >> 4) * 512;
  if (tid >= total) return;
  int frag   = tid >> 9;
  int within = tid & 511;
  int lane = within >> 4;
  int j    = within & 15;
  int ncCount = N >> 4;
  int kc = frag / ncCount;
  int nc = frag - kc * ncCount;
  int n = nc * 16 + (lane & 15);
  int k = kc * 32 + ((lane >> 4) << 3) + (j & 7) + ((j >> 3) << 4);
  float v = (k < Kvalid) ? src[k * N + n] : 0.0f;
  dst[tid] = (_Float16)v;
}

__global__ void cvt_h16(const float* __restrict__ src, _Float16* __restrict__ dst, int n) {
  int t = blockIdx.x * blockDim.x + threadIdx.x;
  if (t < n) dst[t] = (_Float16)src[t];
}

// ---------------- per-node GEMM: Pi = h@W1a, Pj = h@W1b  ([1024,128]@[128,256]) ----------------
__global__ __launch_bounds__(32) void node_gemm(
    const _Float16* __restrict__ h16,
    const _Float16* __restrict__ w1asw, const _Float16* __restrict__ w1bsw,
    float* __restrict__ Pi, float* __restrict__ Pj) {
  const int bx = blockIdx.x;            // 64 M-tiles * 16 N-chunks
  const int mt = bx >> 4, nc = bx & 15;
  const int lane = threadIdx.x & 31, l15 = lane & 15, hf = lane >> 4;
  v8f accA = {}, accB = {};
#pragma unroll
  for (int kc = 0; kc < 4; ++kc) {
    v16h a  = load_a16(h16, 128, mt * 16 + l15, kc * 32, lane);
    v16h ba = load_bfrag(w1asw, kc * 16 + nc, lane);
    v16h bb = load_bfrag(w1bsw, kc * 16 + nc, lane);
    accA = __builtin_amdgcn_wmma_f32_16x16x32_f16(false, a, false, ba, (short)0, accA, false, false);
    accB = __builtin_amdgcn_wmma_f32_16x16x32_f16(false, a, false, bb, (short)0, accB, false, false);
  }
  const int f = nc * 16 + l15;
#pragma unroll
  for (int r = 0; r < 8; ++r) {
    int nd = mt * 16 + r + hf * 8;
    Pi[nd * 256 + f] = accA[r];
    Pj[nd * 256 + f] = accB[r];
  }
}

// ---------------- fused edge kernel: one block per (b,i), 4 waves x 8 j-tiles ----------------
__global__ __launch_bounds__(128) void edge_kernel(
    const float* __restrict__ pos,
    const float* __restrict__ Pi, const float* __restrict__ Pj,
    const _Float16* __restrict__ w1gsw, const _Float16* __restrict__ ew2sw,
    const _Float16* __restrict__ cw1sw,
    const float* __restrict__ eb1, const float* __restrict__ eb2,
    const float* __restrict__ cb1, const float* __restrict__ cw2v,
    const float* __restrict__ cb2,
    float* __restrict__ msg, float* __restrict__ upd) {
  __shared__ _Float16 sX[4][16 * 264];   // post-SiLU edge activations, padded stride
  __shared__ _Float16 sE[4][16 * 136];   // post-cutoff messages (f16)
  __shared__ _Float16 sG[4][16 * 40];    // 17 geometry features padded to K=32
  __shared__ float sCut[4][16], sPdx[4][16], sPdy[4][16];
  __shared__ float sPi[256];

  const int node = blockIdx.x;              // b*512 + i
  const int nodeBase = node & ~(NN - 1);    // b*512
  const int tid = threadIdx.x;
  const int w = tid >> 5, lane = tid & 31, l15 = lane & 15, hf = lane >> 4;

  for (int t = tid; t < 256; t += 128) sPi[t] = Pi[node * 256 + t] + eb1[t];
  __syncthreads();

  const float pix = pos[node * 2 + 0], piy = pos[node * 2 + 1];
  const float cb2s = cb2[0];

  for (int jt = w; jt < 32; jt += 4) {
    const int j0 = jt * 16;
    // ---- geometry: lanes 0-15, one edge each ----
    if (lane < 16) {
      const int j = nodeBase + j0 + lane;
      float dx = pix - pos[j * 2 + 0];
      float dy = piy - pos[j * 2 + 1];
      float d  = sqrtf(dx * dx + dy * dy);
      float inv = 1.0f / (d + 1e-8f);
      float c1 = dx * inv, s1 = dy * inv;     // (cos t, sin t)
      sPdx[w][lane] = dx; sPdy[w][lane] = dy;
      sCut[w][lane] = (d < 10.0f) ? 0.5f * (__cosf(0.3141592653589793f * d) + 1.0f) : 0.0f;
      _Float16* g = &sG[w][lane * 40];
#pragma unroll
      for (int k = 0; k < 8; ++k) {           // gaussian radial basis
        float c = (10.0f / 7.0f) * k;
        float t = (d - c) * 0.8f;             // / width (=1.25)
        g[k] = (_Float16)__expf(-t * t);
      }
      g[8] = (_Float16)1.0f;                  // angular m=0
      float c2 = c1 * c1 - s1 * s1, s2 = 2.0f * c1 * s1;
      float c3 = c2 * c1 - s2 * s1, s3 = s2 * c1 + c2 * s1;
      float c4 = c3 * c1 - s3 * s1, s4 = s3 * c1 + c3 * s1;
      g[9]  = (_Float16)c1; g[10] = (_Float16)c2; g[11] = (_Float16)c3; g[12] = (_Float16)c4;
      g[13] = (_Float16)s1; g[14] = (_Float16)s2; g[15] = (_Float16)s3; g[16] = (_Float16)s4;
#pragma unroll
      for (int k = 17; k < 32; ++k) g[k] = (_Float16)0.0f;
    }
    asm volatile("s_wait_dscnt 0x0" ::: "memory");

    // ---- X = silu(Pi + Pj + G @ Wgeo) : [16 edges, 256] ----
    v16h aG = load_a16(&sG[w][0], 40, l15, 0, lane);
    for (int fc = 0; fc < 16; ++fc) {
      v8f acc = {};
      v16h bG = load_bfrag(w1gsw, fc, lane);
      acc = __builtin_amdgcn_wmma_f32_16x16x32_f16(false, aG, false, bG, (short)0, acc, false, false);
      const int f = fc * 16 + l15;
      const float piF = sPi[f];
#pragma unroll
      for (int r = 0; r < 8; ++r) {
        int m = r + hf * 8;                    // edge row of this D element
        float x = acc[r] + piF + Pj[(nodeBase + j0 + m) * 256 + f];
        sX[w][m * 264 + f] = (_Float16)silu_f(x);
      }
    }
    asm volatile("s_wait_dscnt 0x0" ::: "memory");

    // ---- e = (X @ ew2 + b2) * cutoff : [16,128]; also column-sum into msg_agg ----
    for (int nc = 0; nc < 8; ++nc) {
      v8f acc = {};
#pragma unroll
      for (int kc = 0; kc < 8; ++kc) {
        v16h a = load_a16(&sX[w][0], 264, l15, kc * 32, lane);
        v16h b = load_bfrag(ew2sw, kc * 8 + nc, lane);
        acc = __builtin_amdgcn_wmma_f32_16x16x32_f16(false, a, false, b, (short)0, acc, false, false);
      }
      const int hI = nc * 16 + l15;
      const float bias = eb2[hI];
      float msum = 0.0f;
#pragma unroll
      for (int r = 0; r < 8; ++r) {
        int m = r + hf * 8;
        float e = (acc[r] + bias) * sCut[w][m];
        msum += e;
        sE[w][m * 136 + hI] = (_Float16)e;
      }
      msum += __shfl_xor(msum, 16, 32);
      if (lane < 16) atomicAdd(&msg[node * HID + hI], msum);
    }
    asm volatile("s_wait_dscnt 0x0" ::: "memory");

    // ---- coord weights: cw = silu(e@cw1 + cb1) @ cw2 + cb2 ; upd += pos_diff * cw ----
    float rowacc[8];
#pragma unroll
    for (int r = 0; r < 8; ++r) rowacc[r] = 0.0f;
    for (int nc = 0; nc < 8; ++nc) {
      v8f acc = {};
#pragma unroll
      for (int kc = 0; kc < 4; ++kc) {
        v16h a = load_a16(&sE[w][0], 136, l15, kc * 32, lane);
        v16h b = load_bfrag(cw1sw, kc * 8 + nc, lane);
        acc = __builtin_amdgcn_wmma_f32_16x16x32_f16(false, a, false, b, (short)0, acc, false, false);
      }
      const int hI = nc * 16 + l15;
      const float b1v = cb1[hI], w2v = cw2v[hI];
#pragma unroll
      for (int r = 0; r < 8; ++r) rowacc[r] += silu_f(acc[r] + b1v) * w2v;
    }
    float ux = 0.0f, uy = 0.0f;
#pragma unroll
    for (int r = 0; r < 8; ++r) {
      float v = rowacc[r];
      v += __shfl_xor(v, 1, 32);
      v += __shfl_xor(v, 2, 32);
      v += __shfl_xor(v, 4, 32);
      v += __shfl_xor(v, 8, 32);
      float cwv = v + cb2s;
      int m = r + hf * 8;
      ux += sPdx[w][m] * cwv;
      uy += sPdy[w][m] * cwv;
    }
    if (l15 == 0) {
      ux += __shfl_xor(ux, 16, 32);
      uy += __shfl_xor(uy, 16, 32);
      if (lane == 0) {
        atomicAdd(&upd[node * 2 + 0], ux);
        atomicAdd(&upd[node * 2 + 1], uy);
      }
    }
  }
}

// ---------------- node update + LayerNorm + pos/vel out ----------------
__global__ __launch_bounds__(128) void node_update(
    const float* __restrict__ h, const float* __restrict__ msg, const float* __restrict__ upd,
    const float* __restrict__ nw1, const float* __restrict__ nb1,
    const float* __restrict__ nw2, const float* __restrict__ nb2,
    const float* __restrict__ ln_g, const float* __restrict__ ln_b,
    const float* __restrict__ pos, const float* __restrict__ vel,
    float* __restrict__ out) {
  __shared__ float ni[256];
  __shared__ float hd[128];
  __shared__ float red[128];
  const int node = blockIdx.x, t = threadIdx.x;
  ni[t]       = h[node * 128 + t];
  ni[128 + t] = msg[node * 128 + t];
  __syncthreads();
  float acc = nb1[t];
  for (int k = 0; k < 256; ++k) acc += ni[k] * nw1[k * 128 + t];
  hd[t] = silu_f(acc);
  __syncthreads();
  float acc2 = nb2[t];
  for (int k = 0; k < 128; ++k) acc2 += hd[k] * nw2[k * 128 + t];
  float x = h[node * 128 + t] + acc2;
  red[t] = x; __syncthreads();
  for (int s = 64; s > 0; s >>= 1) { if (t < s) red[t] += red[t + s]; __syncthreads(); }
  float mu = red[0] * (1.0f / 128.0f);
  __syncthreads();
  float dxv = x - mu;
  red[t] = dxv * dxv; __syncthreads();
  for (int s = 64; s > 0; s >>= 1) { if (t < s) red[t] += red[t + s]; __syncthreads(); }
  float var = red[0] * (1.0f / 128.0f);
  float rstd = rsqrtf(var + 1e-5f);
  out[node * 128 + t] = dxv * rstd * ln_g[t] + ln_b[t];
  if (t < 2) {
    float u = upd[node * 2 + t];
    out[131072 + node * 2 + t] = pos[node * 2 + t] + u;          // pos_out
    out[133120 + node * 2 + t] = vel[node * 2 + t] + 0.1f * u;   // vel_out
  }
}

extern "C" void kernel_launch(void* const* d_in, const int* in_sizes, int n_in,
                              void* d_out, int out_size, void* d_ws, size_t ws_size,
                              hipStream_t stream) {
  (void)in_sizes; (void)n_in; (void)out_size; (void)ws_size;
  const float* h   = (const float*)d_in[0];
  const float* pos = (const float*)d_in[1];
  const float* vel = (const float*)d_in[2];
  const float* ew1 = (const float*)d_in[3];
  const float* eb1 = (const float*)d_in[4];
  const float* ew2 = (const float*)d_in[5];
  const float* eb2 = (const float*)d_in[6];
  const float* cw1 = (const float*)d_in[7];
  const float* cb1 = (const float*)d_in[8];
  const float* cw2 = (const float*)d_in[9];
  const float* cb2 = (const float*)d_in[10];
  const float* nw1 = (const float*)d_in[11];
  const float* nb1 = (const float*)d_in[12];
  const float* nw2 = (const float*)d_in[13];
  const float* nb2 = (const float*)d_in[14];
  const float* lng = (const float*)d_in[15];
  const float* lnb = (const float*)d_in[16];
  float* out = (float*)d_out;

  char* ws = (char*)d_ws;
  _Float16* h16   = (_Float16*)(ws + 0);         // 1024*128 f16          (262144 B)
  _Float16* w1asw = (_Float16*)(ws + 262144);    // ew1 rows   0..127     ( 65536 B)
  _Float16* w1bsw = (_Float16*)(ws + 327680);    // ew1 rows 128..255     ( 65536 B)
  _Float16* w1gsw = (_Float16*)(ws + 393216);    // ew1 rows 256..272 ->K=32 (16384 B)
  _Float16* ew2sw = (_Float16*)(ws + 409600);    // 256x128               ( 65536 B)
  _Float16* cw1sw = (_Float16*)(ws + 475136);    // 128x128               ( 32768 B)
  float* Pi  = (float*)(ws + 507904);            // 1024*256 f32          (1 MiB)
  float* Pj  = (float*)(ws + 1556480);           // 1024*256 f32          (1 MiB)
  float* msg = (float*)(ws + 2605056);           // 1024*128 f32          (512 KiB)
  float* upd = (float*)(ws + 3129344);           // 1024*2   f32          (  8 KiB)

  // weight prep (cheap; redone every launch for determinism)
  swizzle_w16<<<128, 256, 0, stream>>>(ew1,             w1asw, 128, 256, 128);
  swizzle_w16<<<128, 256, 0, stream>>>(ew1 + 128 * 256, w1bsw, 128, 256, 128);
  swizzle_w16<<< 32, 256, 0, stream>>>(ew1 + 256 * 256, w1gsw,  32, 256,  17);
  swizzle_w16<<<128, 256, 0, stream>>>(ew2,             ew2sw, 256, 128, 256);
  swizzle_w16<<< 64, 256, 0, stream>>>(cw1,             cw1sw, 128, 128, 128);
  cvt_h16<<<512, 256, 0, stream>>>(h, h16, NNODE * HID);
  hipMemsetAsync(msg, 0, 524288 + 8192, stream);   // msg + upd accumulators

  node_gemm<<<NNODE, 32, 0, stream>>>(h16, w1asw, w1bsw, Pi, Pj);
  edge_kernel<<<NNODE, 128, 0, stream>>>(pos, Pi, Pj, w1gsw, ew2sw, cw1sw,
                                         eb1, eb2, cb1, cw2, cb2, msg, upd);
  node_update<<<NNODE, 128, 0, stream>>>(h, msg, upd, nw1, nb1, nw2, nb2,
                                         lng, lnb, pos, vel, out);
}
